// NPairLoss_40200893890726
// MI455X (gfx1250) — compile-verified, matrix operand order checked by reading, not run on hardware
//
#include <hip/hip_runtime.h>
#include <cmath>

// ---------------- constants ----------------
constexpr int kN    = 16384;        // N_CLASSES
constexpr int kD    = 128;          // feature dim
constexpr int kTile = 16;           // WMMA tile
constexpr int kNT   = kN / kTile;   // 1024 tiles per dimension
constexpr int kMI   = 2;            // i-tiles register-blocked per wave
constexpr int kWavesY = 8;          // waves per block (each owns kMI i-tiles)
constexpr int kJSplit = 16;         // j-dimension split across blocks
constexpr int kJPerBlock = kNT / kJSplit;   // 64 j-tiles, shared by all waves in block

typedef _Float16 f16;
typedef __attribute__((ext_vector_type(16))) _Float16 v16h;
typedef __attribute__((ext_vector_type(8)))  _Float16 v8h;
typedef __attribute__((ext_vector_type(8)))  float    v8f;

static __device__ __forceinline__ v16h join8(v8h a, v8h b) {
    return __builtin_shufflevector(a, b, 0, 1, 2, 3, 4, 5, 6, 7,
                                         8, 9, 10, 11, 12, 13, 14, 15);
}

// ---------------- prep: split-f16 pack + ap dot + zero sums ----------------
// One wave per class i: anchors -> Ahi/Alo, negs -> Bhi/Blo (row-major f16),
// ap[i] = dot(anchor_i, positive_i) in f32, sums[i] = 0.
__global__ __launch_bounds__(256)
void npair_prep(const float* __restrict__ x, const int* __restrict__ aidx,
                f16* __restrict__ Ahi, f16* __restrict__ Alo,
                f16* __restrict__ Bhi, f16* __restrict__ Blo,
                float* __restrict__ ap, float* __restrict__ sums)
{
    const int wave = (blockIdx.x * 256 + threadIdx.x) >> 5;
    const int lane = threadIdx.x & 31;
    if (wave >= kN) return;
    const int i = wave;
    const int a = aidx[i] & 1;
    const float* xa = x + (size_t)(2 * i + a) * kD;        // anchor row
    const float* xp = x + (size_t)(2 * i + (1 - a)) * kD;  // positive row
    const float* xn = x + (size_t)(2 * i + 1) * kD;        // negative row
    float dot = 0.0f;
#pragma unroll
    for (int k = lane; k < kD; k += 32) {
        const float av = xa[k];
        const float nv = xn[k];
        const f16 ah = (f16)av;
        const f16 nh = (f16)nv;
        Ahi[(size_t)i * kD + k] = ah;
        Alo[(size_t)i * kD + k] = (f16)(av - (float)ah);
        Bhi[(size_t)i * kD + k] = nh;
        Blo[(size_t)i * kD + k] = (f16)(nv - (float)nh);
        dot += av * xp[k];
    }
#pragma unroll
    for (int off = 16; off >= 1; off >>= 1)
        dot += __shfl_xor(dot, off, 32);
    if (lane == 0) { ap[i] = dot; sums[i] = 0.0f; }
}

// ---------------- main: WMMA GEMM tiles + exp + partial row sums ----------------
// All 8 waves of a block walk the SAME j-tile stream (WGP$ reuse of B), each
// wave owns kMI i-tiles held entirely in VGPRs.
// S tile = Ahi*Bhi + Ahi*Blo + Alo*Bhi  (f16-split ~ f32 precision).
__global__ __launch_bounds__(32 * kWavesY)
void npair_gemm(const f16* __restrict__ Ahi, const f16* __restrict__ Alo,
                const f16* __restrict__ Bhi, const f16* __restrict__ Blo,
                const float* __restrict__ ap, float* __restrict__ sums)
{
    const int lane = threadIdx.x;      // 0..31
    const int nc   = lane & 15;        // N (and load-row) index within tile
    const int hf   = lane >> 4;        // K-half selector per ISA striping
    const int itBase = (blockIdx.x * kWavesY + threadIdx.y) * kMI;
    const int jt0 = blockIdx.y * kJPerBlock;
    const int jt1 = jt0 + kJPerBlock;

    v16h aHi[kMI][4], aLo[kMI][4];
    float apv[kMI][8];
    float rs[kMI][8];
#pragma unroll
    for (int mi = 0; mi < kMI; ++mi) {
        const int i0 = (itBase + mi) * kTile;
        const size_t arow = (size_t)(i0 + nc) * kD;
#pragma unroll
        for (int kb = 0; kb < 4; ++kb) {
            const int k0 = kb * 32 + 8 * hf;   // lane's K slice: [k0,k0+8) and [k0+16,k0+24)
            aHi[mi][kb] = join8(*(const v8h*)(Ahi + arow + k0),
                                *(const v8h*)(Ahi + arow + k0 + 16));
            aLo[mi][kb] = join8(*(const v8h*)(Alo + arow + k0),
                                *(const v8h*)(Alo + arow + k0 + 16));
        }
#pragma unroll
        for (int r = 0; r < 8; ++r) {
            apv[mi][r] = ap[i0 + r + 8 * hf];  // C/D layout: lane holds M = r + 8*hf
            rs[mi][r]  = 0.0f;
        }
    }

    for (int jt = jt0; jt < jt1; ++jt) {
        const int j0 = jt * kTile;
        const size_t brow = (size_t)(j0 + nc) * kD;   // B column nc = negs row j0+nc

        // Warm WGP$ with the next tile's B rows (one wave per block; 32 lanes
        // cover 16 rows x two 128B halves). Lowers to global_prefetch_b8.
        if (threadIdx.y == 0 && jt + 1 < jt1) {
            const size_t prow = (size_t)((jt + 1) * kTile + nc) * kD + hf * 64;
            __builtin_prefetch(Bhi + prow, 0, 3);
            __builtin_prefetch(Blo + prow, 0, 3);
        }

        v8f acc[kMI];
#pragma unroll
        for (int mi = 0; mi < kMI; ++mi)
            acc[mi] = v8f{0.f, 0.f, 0.f, 0.f, 0.f, 0.f, 0.f, 0.f};

#pragma unroll
        for (int kb = 0; kb < 4; ++kb) {
            const int k0 = kb * 32 + 8 * hf;
            const v16h bHi = join8(*(const v8h*)(Bhi + brow + k0),
                                   *(const v8h*)(Bhi + brow + k0 + 16));
            const v16h bLo = join8(*(const v8h*)(Blo + brow + k0),
                                   *(const v8h*)(Blo + brow + k0 + 16));
#pragma unroll
            for (int mi = 0; mi < kMI; ++mi) {
                acc[mi] = __builtin_amdgcn_wmma_f32_16x16x32_f16(
                    false, aHi[mi][kb], false, bHi, (short)0, acc[mi], false, false);
                acc[mi] = __builtin_amdgcn_wmma_f32_16x16x32_f16(
                    false, aHi[mi][kb], false, bLo, (short)0, acc[mi], false, false);
                acc[mi] = __builtin_amdgcn_wmma_f32_16x16x32_f16(
                    false, aLo[mi][kb], false, bHi, (short)0, acc[mi], false, false);
            }
        }

#pragma unroll
        for (int mi = 0; mi < kMI; ++mi) {
            const int i0 = (itBase + mi) * kTile;
            const bool diagTile = (j0 == i0);
#pragma unroll
            for (int r = 0; r < 8; ++r) {
                float e = __expf(acc[mi][r] - apv[mi][r]);
                if (diagTile && (r + 8 * hf) == nc) e = 0.0f;  // mask j == i
                rs[mi][r] += e;
            }
        }
    }

    // Reduce across the 16-lane N groups, then one atomic per row.
#pragma unroll
    for (int mi = 0; mi < kMI; ++mi) {
        const int i0 = (itBase + mi) * kTile;
#pragma unroll
        for (int r = 0; r < 8; ++r) {
            float v = rs[mi][r];
            v += __shfl_xor(v, 1, 16);
            v += __shfl_xor(v, 2, 16);
            v += __shfl_xor(v, 4, 16);
            v += __shfl_xor(v, 8, 16);
            if (nc == 0) atomicAdd(&sums[i0 + r + 8 * hf], v);
        }
    }
}

// ---------------- finalize: sum log1p(sums) -> scalar ----------------
__global__ __launch_bounds__(256)
void npair_finalize(const float* __restrict__ sums, float* __restrict__ out)
{
    __shared__ float sh[256];
    float s = 0.0f;
    for (int i = threadIdx.x; i < kN; i += 256)
        s += log1pf(sums[i]);
    sh[threadIdx.x] = s;
    __syncthreads();
    for (int w = 128; w > 0; w >>= 1) {
        if ((int)threadIdx.x < w) sh[threadIdx.x] += sh[threadIdx.x + w];
        __syncthreads();
    }
    if (threadIdx.x == 0) out[0] = sh[0];
}

// ---------------- launch ----------------
extern "C" void kernel_launch(void* const* d_in, const int* in_sizes, int n_in,
                              void* d_out, int out_size, void* d_ws, size_t ws_size,
                              hipStream_t stream)
{
    (void)in_sizes; (void)n_in; (void)out_size; (void)ws_size;
    const float* x    = (const float*)d_in[0];
    const int*   aidx = (const int*)d_in[1];
    // d_in[2] (pos_idx) is implied by 1 - anchor_idx; unused.

    char* ws = (char*)d_ws;
    const size_t matBytes = (size_t)kN * kD * sizeof(f16);   // 4 MiB each
    f16*   Ahi  = (f16*)(ws + 0 * matBytes);
    f16*   Alo  = (f16*)(ws + 1 * matBytes);
    f16*   Bhi  = (f16*)(ws + 2 * matBytes);
    f16*   Blo  = (f16*)(ws + 3 * matBytes);
    float* ap   = (float*)(ws + 4 * matBytes);
    float* sums = (float*)(ws + 4 * matBytes + (size_t)kN * sizeof(float));

    npair_prep<<<dim3(kN / 8), dim3(256), 0, stream>>>(x, aidx, Ahi, Alo, Bhi, Blo, ap, sums);

    npair_gemm<<<dim3(kNT / (kMI * kWavesY), kJSplit), dim3(32, kWavesY), 0, stream>>>(
        Ahi, Alo, Bhi, Blo, ap, sums);

    npair_finalize<<<1, 256, 0, stream>>>(sums, (float*)d_out);
}
